// MFF_MoE_20641612825085
// MI455X (gfx1250) — matrix operand order repaired and checked by reading, not compiled, not run on hardware
//
#include <hip/hip_runtime.h>
#include <math.h>
#include <stdint.h>

// ---------------- problem constants (match reference) ----------------
static constexpr int kT   = 8192;   // tokens
static constexpr int kD   = 1024;   // d_model
static constexpr int kH   = 4096;   // d_ff
static constexpr int kE   = 7;      // experts
static constexpr int kK   = 2;      // top-k
static constexpr int kCAP = 3072;   // per-expert capacity

// ---------------- WMMA vector types (CDNA5, wave32) ----------------
typedef __attribute__((ext_vector_type(16))) __bf16 v16bf;
typedef __attribute__((ext_vector_type(8)))  __bf16 v8bf;
typedef __attribute__((ext_vector_type(8)))  float  v8f;

// ---------------- GEMM tiling ----------------
static constexpr int kBM = 128;  // block rows
static constexpr int kBN = 64;   // block cols
static constexpr int kBK = 64;   // staged k-step = 2x WMMA-K (bf16 K=32)

// =====================================================================
// 1) Router: gates = softmax(x @ Wg); top-2; renormalize.
//    One wave32 per token. E=7 accumulators per lane, shfl reduction.
// =====================================================================
__global__ __launch_bounds__(256) void router_kernel(
    const float* __restrict__ x, const float* __restrict__ Wg,
    int* __restrict__ flat_e, float* __restrict__ flat_w) {
  int gid   = blockIdx.x * blockDim.x + threadIdx.x;
  int token = gid >> 5;
  int lane  = threadIdx.x & 31;
  if (token >= kT) return;

  float acc[kE];
#pragma unroll
  for (int e = 0; e < kE; ++e) acc[e] = 0.f;

  const float* xr = x + (size_t)token * kD;
  for (int d = lane; d < kD; d += 32) {
    float xv = xr[d];
#pragma unroll
    for (int e = 0; e < kE; ++e) acc[e] += xv * Wg[d * kE + e];
  }
#pragma unroll
  for (int e = 0; e < kE; ++e)
    for (int off = 16; off > 0; off >>= 1)
      acc[e] += __shfl_xor(acc[e], off, 32);

  if (lane == 0) {
    int i1 = 0;
    for (int e = 1; e < kE; ++e) if (acc[e] > acc[i1]) i1 = e;
    int i2 = (i1 == 0) ? 1 : 0;
    for (int e = 0; e < kE; ++e) if (e != i1 && acc[e] > acc[i2]) i2 = e;
    // top-k-of-softmax renormalized == softmax over the two top logits
    float w1 = 1.f / (1.f + __expf(acc[i2] - acc[i1]));
    flat_e[2 * token]     = i1;
    flat_e[2 * token + 1] = i2;
    flat_w[2 * token]     = w1;
    flat_w[2 * token + 1] = 1.f - w1;
  }
}

// =====================================================================
// 2) generic 32-bit fill
// =====================================================================
__global__ void fill_u32(unsigned* __restrict__ p, unsigned v, int n) {
  int i = blockIdx.x * blockDim.x + threadIdx.x;
  if (i < n) p[i] = v;
}

// =====================================================================
// 3) Slot assignment, token-order priority (deterministic prefix scan,
//    single block). Builds inverse map slot -> flat entry (token*2+k).
// =====================================================================
__global__ __launch_bounds__(256) void slot_kernel(
    const int* __restrict__ flat_e, const float* __restrict__ flat_w,
    int* __restrict__ tok_of_slot, float* __restrict__ w_of_slot) {
  __shared__ int sh_e[256];
  __shared__ int base[kE];
  int tid = threadIdx.x;
  if (tid < kE) base[tid] = 0;
  __syncthreads();

  const int TK = kT * kK;  // 16384, multiple of 256
  for (int c = 0; c < TK; c += 256) {
    int i = c + tid;
    int e = flat_e[i];
    sh_e[tid] = e;
    __syncthreads();
    int pos = 0;
    for (int j = 0; j < tid; ++j) pos += (sh_e[j] == e);
    int g = base[e] + pos;  // global position within expert (token order)
    if (g < kCAP) {
      tok_of_slot[e * kCAP + g] = i;          // encodes token=i>>1, k=i&1
      w_of_slot[e * kCAP + g]   = flat_w[i];
    }
    __syncthreads();
    if (tid < kE) {
      int cnt = 0;
      for (int j = 0; j < 256; ++j) cnt += (sh_e[j] == tid);
      base[tid] += cnt;
    }
    __syncthreads();
  }
}

// =====================================================================
// 4) Dispatch: gather x rows into bf16 [E,CAP,D] buffer.
//    Unfilled slots keep garbage; their results are never stored.
// =====================================================================
__global__ __launch_bounds__(256) void dispatch_kernel(
    const float* __restrict__ x, const int* __restrict__ tok_of_slot,
    __bf16* __restrict__ bufB) {
  int row = blockIdx.x;  // 0 .. E*CAP-1
  int ti = tok_of_slot[row];
  if (ti < 0) return;
  int token = ti >> 1;
  const float* src = x + (size_t)token * kD;
  __bf16* dst = bufB + (size_t)row * kD;
  for (int d = threadIdx.x; d < kD; d += blockDim.x)
    dst[d] = (__bf16)src[d];
}

// =====================================================================
// Helpers
// =====================================================================

// branch-free gelu(tanh approx): tanh(u) = sign(u)*(1-t)/(1+t), t=e^{-2|u|}
__device__ __forceinline__ float gelu_tanh(float v) {
  float u  = 0.7978845608028654f * (v + 0.044715f * v * v * v);
  float t  = __expf(-2.f * fabsf(u));
  float th = __builtin_copysignf((1.f - t) / (1.f + t), u);
  return 0.5f * v * (1.f + th);
}

// async memory->LDS copy of 16 bytes (CDNA5 path, tracked by ASYNCcnt)
__device__ __forceinline__ void async_copy_b128(const void* gptr, void* lptr) {
  unsigned           ldsa = (unsigned)(uintptr_t)lptr;       // low 32b = LDS addr
  unsigned long long ga   = (unsigned long long)(uintptr_t)gptr;
  asm volatile("global_load_async_to_lds_b128 %0, %1, off"
               :: "v"(ldsa), "v"(ga) : "memory");
}
__device__ __forceinline__ void wait_async() {
  asm volatile("s_wait_asynccnt 0x0" ::: "memory");
}

// A 16x32 bf16 fragment: lanes 0-15 row M=lane, K {0..7,16..23};
// lanes 16-31 same row, K {8..15,24..31} (ISA 7.12.2 interleave).
__device__ __forceinline__ v16bf load_a_frag(const __bf16* As, int ldk,
                                             int rowBase, int lane) {
  int row = rowBase + (lane & 15);
  int ko  = (lane >> 4) * 8;
  const __bf16* p = As + row * ldk;
  v16bf a;
#pragma unroll
  for (int i = 0; i < 8; ++i) {
    a[i]     = p[ko + i];
    a[i + 8] = p[16 + ko + i];
  }
  return a;
}

// B 32x16 bf16 fragment (Bs stored col-major [n][k]): lanes 0-15 col N=lane,
// K {0..7,16..23}; lanes 16-31 col N=lane-16, K {8..15,24..31}.
__device__ __forceinline__ v16bf load_b_frag(const __bf16* Bs, int ldk,
                                             int colBase, int lane) {
  int col = colBase + (lane & 15);
  int kb  = (lane >> 4) * 8;
  const __bf16* p = Bs + col * ldk + kb;
  v16bf b;
#pragma unroll
  for (int i = 0; i < 8; ++i) {
    b[i]     = p[i];
    b[i + 8] = p[16 + i];
  }
  return b;
}

// =====================================================================
// 5) GEMM1: h = gelu(buf[e] @ W1[e]); A bf16 [CAP,D], B fp32->bf16 [D,H].
//    256 threads = 8 waves; wave grid 4x2 over a 128x64 block tile,
//    each wave: 2x2 of 16x16 WMMA, two K=32 sub-steps per staged tile.
//    A staged via global_load_async_to_lds_b128 (ASYNCcnt), B via VGPR
//    with fp32->bf16 convert; next B tile prefetched.
// =====================================================================
__global__ __launch_bounds__(256) void gemm1_kernel(
    const __bf16* __restrict__ bufB, const float* __restrict__ W1,
    __bf16* __restrict__ hB) {
  __shared__ __bf16 As[kBM][kBK];  // 16 KB
  __shared__ __bf16 Bs[kBN][kBK];  // 8 KB, col-major tile [n][k]

  const int e = blockIdx.z;
  const int rowBase = blockIdx.y * kBM;
  const int colBase = blockIdx.x * kBN;
  const __bf16* A = bufB + (size_t)e * kCAP * kD;
  const float*  B = W1   + (size_t)e * kD * kH;
  __bf16*       C = hB   + (size_t)e * kCAP * kH;

  const int tid  = threadIdx.x;
  const int lane = tid & 31;
  const int wave = tid >> 5;
  const int wr = (wave & 3) * 32;   // wave row offset in tile
  const int wc = (wave >> 2) * 32;  // wave col offset in tile

  v8f acc[2][2];
#pragma unroll
  for (int m = 0; m < 2; ++m)
#pragma unroll
    for (int n = 0; n < 2; ++n) acc[m][n] = (v8f)0.f;

  for (int k0 = 0; k0 < kD; k0 += kBK) {
    // A tile 128x64 bf16 = 1024 x 16B: 4 async b128 copies per thread
#pragma unroll
    for (int i = 0; i < 4; ++i) {
      int linear = tid + i * 256;      // 0..1023
      int r  = linear >> 3;            // 0..127
      int kk = (linear & 7) * 8;       // 0..56, 16B-aligned
      async_copy_b128(A + (size_t)(rowBase + r) * kD + k0 + kk, &As[r][kk]);
    }
    // prefetch next B tile into L2 (global_prefetch_b8)
    if (k0 + kBK < kD)
      __builtin_prefetch(&B[(size_t)(k0 + kBK) * kH + colBase + (tid & 63)], 0, 1);
    // B tile 64x64 fp32 -> bf16, transposed into Bs[n][k] (coalesced reads)
#pragma unroll
    for (int i = 0; i < 16; ++i) {
      int linear = tid + i * 256;
      int n  = linear & 63;
      int kk = linear >> 6;
      Bs[n][kk] = (__bf16)B[(size_t)(k0 + kk) * kH + colBase + n];
    }
    wait_async();
    __syncthreads();

#pragma unroll
    for (int ks = 0; ks < kBK; ks += 32) {
      v16bf af[2], bf2[2];
#pragma unroll
      for (int s = 0; s < 2; ++s) {
        af[s]  = load_a_frag(&As[0][0] + ks, kBK, wr + s * 16, lane);
        bf2[s] = load_b_frag(&Bs[0][0] + ks, kBK, wc + s * 16, lane);
      }
#pragma unroll
      for (int m = 0; m < 2; ++m)
#pragma unroll
        for (int n = 0; n < 2; ++n)
          acc[m][n] = __builtin_amdgcn_wmma_f32_16x16x32_bf16(
              false, af[m], false, bf2[n], (short)0, acc[m][n], false, false);
    }
    __syncthreads();
  }

  // epilogue: gelu -> bf16. C layout: VGPR i holds M=i (+8 for lanes>=16).
#pragma unroll
  for (int m = 0; m < 2; ++m)
#pragma unroll
    for (int n = 0; n < 2; ++n) {
      int ncol = colBase + wc + n * 16 + (lane & 15);
      int moff = rowBase + wr + m * 16 + ((lane >> 4) * 8);
#pragma unroll
      for (int i = 0; i < 8; ++i)
        C[(size_t)(moff + i) * kH + ncol] = (__bf16)gelu_tanh(acc[m][n][i]);
    }
}

// =====================================================================
// 6) GEMM2: y = h[e] @ W2[e]; epilogue scatters w*y into
//    out_partial[k][token][:] (one slot -> one (token,k): race-free).
// =====================================================================
__global__ __launch_bounds__(256) void gemm2_kernel(
    const __bf16* __restrict__ hB, const float* __restrict__ W2,
    const int* __restrict__ tok_of_slot, const float* __restrict__ w_of_slot,
    float* __restrict__ out_partial) {
  __shared__ __bf16 As[kBM][kBK];
  __shared__ __bf16 Bs[kBN][kBK];

  const int e = blockIdx.z;
  const int rowBase = blockIdx.y * kBM;
  const int colBase = blockIdx.x * kBN;
  const __bf16* A = hB + (size_t)e * kCAP * kH;
  const float*  B = W2 + (size_t)e * kH * kD;

  const int tid  = threadIdx.x;
  const int lane = tid & 31;
  const int wave = tid >> 5;
  const int wr = (wave & 3) * 32;
  const int wc = (wave >> 2) * 32;

  v8f acc[2][2];
#pragma unroll
  for (int m = 0; m < 2; ++m)
#pragma unroll
    for (int n = 0; n < 2; ++n) acc[m][n] = (v8f)0.f;

  for (int k0 = 0; k0 < kH; k0 += kBK) {
#pragma unroll
    for (int i = 0; i < 4; ++i) {
      int linear = tid + i * 256;
      int r  = linear >> 3;
      int kk = (linear & 7) * 8;
      async_copy_b128(A + (size_t)(rowBase + r) * kH + k0 + kk, &As[r][kk]);
    }
    if (k0 + kBK < kH)
      __builtin_prefetch(&B[(size_t)(k0 + kBK) * kD + colBase + (tid & 63)], 0, 1);
#pragma unroll
    for (int i = 0; i < 16; ++i) {
      int linear = tid + i * 256;
      int n  = linear & 63;
      int kk = linear >> 6;
      Bs[n][kk] = (__bf16)B[(size_t)(k0 + kk) * kD + colBase + n];
    }
    wait_async();
    __syncthreads();

#pragma unroll
    for (int ks = 0; ks < kBK; ks += 32) {
      v16bf af[2], bf2[2];
#pragma unroll
      for (int s = 0; s < 2; ++s) {
        af[s]  = load_a_frag(&As[0][0] + ks, kBK, wr + s * 16, lane);
        bf2[s] = load_b_frag(&Bs[0][0] + ks, kBK, wc + s * 16, lane);
      }
#pragma unroll
      for (int m = 0; m < 2; ++m)
#pragma unroll
        for (int n = 0; n < 2; ++n)
          acc[m][n] = __builtin_amdgcn_wmma_f32_16x16x32_bf16(
              false, af[m], false, bf2[n], (short)0, acc[m][n], false, false);
    }
    __syncthreads();
  }

  // epilogue: slot -> (token, k, weight); dropped/unfilled slots skipped
#pragma unroll
  for (int m = 0; m < 2; ++m) {
    int mbase = rowBase + wr + m * 16 + ((lane >> 4) * 8);
#pragma unroll
    for (int i = 0; i < 8; ++i) {
      int row = mbase + i;                      // slot within expert
      int ti  = tok_of_slot[e * kCAP + row];
      if (ti < 0) continue;
      int   token = ti >> 1;
      int   kk    = ti & 1;
      float w     = w_of_slot[e * kCAP + row];
      float* dst  = out_partial + ((size_t)kk * kT + token) * kD;
#pragma unroll
      for (int n = 0; n < 2; ++n) {
        int ncol = colBase + wc + n * 16 + (lane & 15);
        dst[ncol] = w * acc[m][n][i];
      }
    }
  }
}

// =====================================================================
// 7) Combine: out = partial[k=0] + partial[k=1]
// =====================================================================
__global__ void combine_kernel(const float* __restrict__ p,
                               float* __restrict__ out) {
  int i = blockIdx.x * blockDim.x + threadIdx.x;
  if (i < kT * kD) out[i] = p[i] + p[(size_t)kT * kD + i];
}

// =====================================================================
// Host-side launch
// =====================================================================
extern "C" void kernel_launch(void* const* d_in, const int* in_sizes, int n_in,
                              void* d_out, int out_size, void* d_ws, size_t ws_size,
                              hipStream_t stream) {
  (void)in_sizes; (void)n_in; (void)out_size; (void)ws_size;
  const float* x  = (const float*)d_in[0];
  const float* Wg = (const float*)d_in[1];
  const float* W1 = (const float*)d_in[2];
  const float* W2 = (const float*)d_in[3];
  float* out = (float*)d_out;

  // workspace carve-out (~288 MB): router arrays + bf16 dispatch buf +
  // bf16 h + fp32 out_partial
  char* ws = (char*)d_ws;
  size_t off = 0;
  auto carve = [&](size_t bytes) {
    size_t o = off;
    off += (bytes + 255) & ~(size_t)255;
    return o;
  };
  int*    flat_e = (int*)   (ws + carve((size_t)kT * kK * 4));
  float*  flat_w = (float*) (ws + carve((size_t)kT * kK * 4));
  int*    tok_of = (int*)   (ws + carve((size_t)kE * kCAP * 4));
  float*  w_of   = (float*) (ws + carve((size_t)kE * kCAP * 4));
  __bf16* bufB   = (__bf16*)(ws + carve((size_t)kE * kCAP * kD * 2));
  __bf16* hB     = (__bf16*)(ws + carve((size_t)kE * kCAP * kH * 2));
  float*  outp   = (float*) (ws + carve((size_t)2 * kT * kD * 4));

  router_kernel<<<kT / 8, 256, 0, stream>>>(x, Wg, flat_e, flat_w);
  fill_u32<<<(kE * kCAP + 255) / 256, 256, 0, stream>>>(
      (unsigned*)tok_of, 0xFFFFFFFFu, kE * kCAP);
  fill_u32<<<(2 * kT * kD + 255) / 256, 256, 0, stream>>>(
      (unsigned*)outp, 0u, 2 * kT * kD);
  slot_kernel<<<1, 256, 0, stream>>>(flat_e, flat_w, tok_of, w_of);
  dispatch_kernel<<<kE * kCAP, 256, 0, stream>>>(x, tok_of, bufB);
  gemm1_kernel<<<dim3(kH / kBN, kCAP / kBM, kE), 256, 0, stream>>>(bufB, W1, hB);
  gemm2_kernel<<<dim3(kD / kBN, kCAP / kBM, kE), 256, 0, stream>>>(
      hB, W2, tok_of, w_of, outp);
  combine_kernel<<<(kT * kD + 255) / 256, 256, 0, stream>>>(outp, out);
}
